// GATI_26216480375126
// MI455X (gfx1250) — matrix operand order repaired and checked by reading, not compiled, not run on hardware
//
#include <hip/hip_runtime.h>
#include <hip/hip_bf16.h>
#include <stdint.h>

#define EMB 1280
#define MDIM 200
#define LN_EPS 1e-5f

typedef __attribute__((ext_vector_type(16))) __bf16 v16bf;
typedef __attribute__((ext_vector_type(8)))  float  v8f;
typedef __attribute__((ext_vector_type(4)))  float  f32x4;

// ---------------- wave helpers (wave32) ----------------
__device__ inline float waveReduceSum(float v) {
    #pragma unroll
    for (int off = 16; off > 0; off >>= 1) v += __shfl_down(v, off, 32);
    return v; // valid in lane 0
}

__device__ inline void atomicMaxFloat(float* addr, float val) {
    // ordered-int trick: works across mixed signs, init = -INF
    if (!(val < 0.0f)) atomicMax((int*)addr, __float_as_int(val));
    else               atomicMin((unsigned int*)addr, __float_as_uint(val));
}

// ---------------- generic fill ----------------
__global__ void k_fill(float* p, size_t n, float v) {
    size_t i = (size_t)blockIdx.x * 256u + threadIdx.x;
    if (i < n) p[i] = v;
}

// ---------------- WMMA bf16 GEMM: C[M,N] = A[M,K] * B[K,N], B given as BT[N,K] ----------------
__device__ inline v16bf load_a_frag(const __bf16* A, int row, int ldk, int k, int half) {
    const __bf16* p = A + (size_t)row * ldk + k + 8 * half;
    v16bf a;
    ((f32x4*)&a)[0] = *(const f32x4*)(p);        // elems 0..7  -> K = k+8h .. +7
    ((f32x4*)&a)[1] = *(const f32x4*)(p + 16);   // elems 8..15 -> K = k+16+8h .. +7
    return a;
}
__device__ inline v16bf load_b_frag(const __bf16* BT, int col, int ldk, int k, int half) {
    const __bf16* p = BT + (size_t)col * ldk + k + 8 * half;
    v16bf b;
    ((f32x4*)&b)[0] = *(const f32x4*)(p);        // elems 0..7  -> K = k+8h .. +7
    ((f32x4*)&b)[1] = *(const f32x4*)(p + 16);   // elems 8..15 -> K = k+16+8h .. +7
    return b;
}

__global__ __launch_bounds__(256)
void k_gemm_bf16(const __bf16* __restrict__ A, const __bf16* __restrict__ BT,
                 float* __restrict__ C, int M, int Nn, int Kd) {
    const int tid  = threadIdx.x;
    const int wave = tid >> 5;
    const int lane = tid & 31;
    const int l    = lane & 15;
    const int hf   = lane >> 4;
    const int wm   = wave & 3;        // 4 waves along M
    const int wn   = wave >> 2;       // 2 waves along N
    const int row0 = blockIdx.y * 128 + wm * 32;
    const int col0 = blockIdx.x * 128 + wn * 64;
    if (row0 >= M) return;            // M % 32 == 0 so wave tiles are all-in or all-out

    v8f acc[2][4];
    const v8f zero = {0.f,0.f,0.f,0.f,0.f,0.f,0.f,0.f};
    #pragma unroll
    for (int i = 0; i < 2; ++i)
        #pragma unroll
        for (int j = 0; j < 4; ++j) acc[i][j] = zero;

    for (int k = 0; k < Kd; k += 32) {
        v16bf af[2], bfr[4];
        #pragma unroll
        for (int i = 0; i < 2; ++i) af[i]  = load_a_frag(A,  row0 + 16*i + l, Kd, k, hf);
        #pragma unroll
        for (int j = 0; j < 4; ++j) bfr[j] = load_b_frag(BT, col0 + 16*j + l, Kd, k, hf);
        #pragma unroll
        for (int i = 0; i < 2; ++i)
            #pragma unroll
            for (int j = 0; j < 4; ++j)
                acc[i][j] = __builtin_amdgcn_wmma_f32_16x16x32_bf16(
                    false, af[i], false, bfr[j], (short)0, acc[i][j], false, false);
    }

    #pragma unroll
    for (int i = 0; i < 2; ++i) {
        const int rbase = row0 + 16*i + 8*hf;
        #pragma unroll
        for (int j = 0; j < 4; ++j) {
            const int c = col0 + 16*j + l;
            #pragma unroll
            for (int r = 0; r < 8; ++r)
                C[(size_t)(rbase + r) * Nn + c] = acc[i][j][r];
        }
    }
}

// ---------------- transpose + convert weights: WT[f,d] = bf16(W[d,f]) ----------------
__global__ void k_wtrans(const float* __restrict__ W, __bf16* __restrict__ WT, int D, int F) {
    int idx = blockIdx.x * 256 + threadIdx.x;
    if (idx >= D * F) return;
    int f = idx / D, d = idx - f * D;
    WT[idx] = (__bf16)W[(size_t)d * F + f];
}

// ---------------- cross-attention (rank-1 collapsed) ----------------
__global__ void k_kv(const float* __restrict__ x,
                     const float* kw, const float* kb,
                     const float* vw, const float* vb, const float* qb,
                     float* Kvec, float* Vvec, float* qbK) {
    int j = threadIdx.x; // 256 threads
    const float* md = x + EMB;
    float ka = kb[j], va = vb[j];
    for (int i = 0; i < MDIM; ++i) {
        float m = md[i];
        ka += m * kw[i * 256 + j];
        va += m * vw[i * 256 + j];
    }
    Kvec[j] = ka; Vvec[j] = va;
    __shared__ float sh[256];
    sh[j] = qb[j] * ka;
    __syncthreads();
    for (int s = 128; s > 0; s >>= 1) { if (j < s) sh[j] += sh[j + s]; __syncthreads(); }
    if (j == 0) qbK[0] = sh[0];
}

__global__ void k_wu(const float* __restrict__ qw, const float* __restrict__ ow,
                     const float* Kvec, const float* Vvec,
                     float* wvec, float* uvec) {
    int i = blockIdx.x * 256 + threadIdx.x;
    if (i >= EMB) return;
    float wa = 0.f, ua = 0.f;
    for (int j = 0; j < 256; ++j) {
        wa += qw[(size_t)i * 256 + j] * Kvec[j];
        ua += Vvec[j] * ow[(size_t)j * EMB + i];
    }
    wvec[i] = wa; uvec[i] = ua;
}

__global__ __launch_bounds__(256)
void k_scores(const float* __restrict__ x, const float* __restrict__ wvec,
              const float* qbK, float* s, int n) {
    int gw   = (blockIdx.x * 256 + threadIdx.x) >> 5;
    int lane = threadIdx.x & 31;
    if (gw >= n) return;
    const float* xe = x + (size_t)gw * (EMB + MDIM);
    float acc = 0.f;
    for (int i = lane; i < EMB; i += 32) acc += xe[i] * wvec[i];
    acc = waveReduceSum(acc);
    if (lane == 0) s[gw] = (acc + qbK[0]) * 0.0625f; // 1/sqrt(256)
}

__global__ void k_pmax(const float* s, int n, float* pmax) {
    __shared__ float sh[256];
    float m = -INFINITY;
    for (int i = blockIdx.x * 256 + threadIdx.x; i < n; i += gridDim.x * 256)
        m = fmaxf(m, s[i]);
    sh[threadIdx.x] = m; __syncthreads();
    for (int st = 128; st > 0; st >>= 1) {
        if (threadIdx.x < st) sh[threadIdx.x] = fmaxf(sh[threadIdx.x], sh[threadIdx.x + st]);
        __syncthreads();
    }
    if (threadIdx.x == 0) pmax[blockIdx.x] = sh[0];
}
__global__ void k_gmax(const float* pmax, float* gmax) { // 1 block, 256 partials
    __shared__ float sh[256];
    sh[threadIdx.x] = pmax[threadIdx.x]; __syncthreads();
    for (int st = 128; st > 0; st >>= 1) {
        if (threadIdx.x < st) sh[threadIdx.x] = fmaxf(sh[threadIdx.x], sh[threadIdx.x + st]);
        __syncthreads();
    }
    if (threadIdx.x == 0) gmax[0] = sh[0];
}
__global__ void k_expsum(const float* s, const float* gmax, int n, float* ex, float* psum) {
    __shared__ float sh[256];
    float g = gmax[0], a = 0.f;
    for (int i = blockIdx.x * 256 + threadIdx.x; i < n; i += gridDim.x * 256) {
        float e = __expf(s[i] - g);
        ex[i] = e; a += e;
    }
    sh[threadIdx.x] = a; __syncthreads();
    for (int st = 128; st > 0; st >>= 1) {
        if (threadIdx.x < st) sh[threadIdx.x] += sh[threadIdx.x + st];
        __syncthreads();
    }
    if (threadIdx.x == 0) psum[blockIdx.x] = sh[0];
}
__global__ void k_gsum(const float* psum, float* gsum) {
    __shared__ float sh[256];
    sh[threadIdx.x] = psum[threadIdx.x]; __syncthreads();
    for (int st = 128; st > 0; st >>= 1) {
        if (threadIdx.x < st) sh[threadIdx.x] += sh[threadIdx.x + st];
        __syncthreads();
    }
    if (threadIdx.x == 0) gsum[0] = sh[0];
}

__global__ void k_h0(const float* __restrict__ x, const float* ob, const float* uvec,
                     const float* ex, const float* gsum,
                     float* __restrict__ h0, __bf16* __restrict__ hb, int n) {
    size_t idx = (size_t)blockIdx.x * 256u + threadIdx.x;
    size_t total = (size_t)n * EMB;
    if (idx >= total) return;
    int node = (int)(idx / EMB);
    int i    = (int)(idx - (size_t)node * EMB);
    float w = ex[node] / gsum[0];
    float v = x[(size_t)node * (EMB + MDIM) + i] + ob[i] + w * uvec[i];
    h0[idx] = v;
    hb[idx] = (__bf16)v;
}

// ---------------- graph degree / self-loop attr (fill='mean') ----------------
__global__ void k_deg(const int* __restrict__ ei, const float* __restrict__ ea, int E_,
                      float* cnt, float* sattr) {
    int e = blockIdx.x * 256 + threadIdx.x;
    if (e >= E_) return;
    int d = ei[E_ + e];
    atomicAdd(&cnt[d], 1.0f);
    atomicAdd(&sattr[d], ea[e]);
}
__global__ void k_loopattr(const float* cnt, const float* sattr, float* loopA, int n) {
    int i = blockIdx.x * 256 + threadIdx.x;
    if (i >= n) return;
    float c = cnt[i];
    loopA[i] = (c > 0.f) ? sattr[i] / fmaxf(c, 1.f) : 0.f;
}

// ---------------- GAT attention pieces ----------------
__global__ __launch_bounds__(256)
void k_aldot(const float* __restrict__ H, const float* as, const float* ad,
             float* al_s, float* al_d, int n, int heads) {
    int gw   = (blockIdx.x * 256 + threadIdx.x) >> 5;
    int lane = threadIdx.x & 31;
    if (gw >= n * heads) return;
    int node = gw / heads, hd = gw - node * heads;
    const float* hp = H + (size_t)node * heads * 256 + hd * 256;
    float s = 0.f, d = 0.f;
    for (int c = lane; c < 256; c += 32) {
        float v = hp[c];
        s += v * as[hd * 256 + c];
        d += v * ad[hd * 256 + c];
    }
    s = waveReduceSum(s); d = waveReduceSum(d);
    if (lane == 0) { al_s[gw] = s; al_d[gw] = d; }
}

__global__ void k_wedot(const float* We, const float* ae, float* wedot, int heads) {
    int wave = threadIdx.x >> 5, lane = threadIdx.x & 31;
    if (wave >= heads) return;
    float a = 0.f;
    for (int c = lane; c < 256; c += 32) a += We[wave * 256 + c] * ae[wave * 256 + c];
    a = waveReduceSum(a);
    if (lane == 0) wedot[wave] = a;
}

__global__ void k_alpha(const int* __restrict__ ei, const float* __restrict__ ea,
                        const float* loopA, const float* al_s, const float* al_d,
                        const float* wedot, float* alpha, float* amax,
                        int E_, int n, int heads) {
    int e = blockIdx.x * 256 + threadIdx.x;
    int tot = E_ + n;
    if (e >= tot) return;
    int src, dst; float a;
    if (e < E_) { src = ei[e]; dst = ei[E_ + e]; a = ea[e]; }
    else        { src = dst = e - E_;            a = loopA[e - E_]; }
    for (int hd = 0; hd < heads; ++hd) {
        float v = al_s[src * heads + hd] + al_d[dst * heads + hd] + a * wedot[hd];
        v = (v > 0.f) ? v : 0.2f * v;             // leaky_relu(0.2)
        alpha[(size_t)e * heads + hd] = v;
        atomicMaxFloat(&amax[dst * heads + hd], v);
    }
}

__global__ void k_exden(const int* __restrict__ ei, const float* __restrict__ alpha,
                        const float* amax, float* exv, float* denom,
                        int E_, int n, int heads) {
    int e = blockIdx.x * 256 + threadIdx.x;
    int tot = E_ + n;
    if (e >= tot) return;
    int dst = (e < E_) ? ei[E_ + e] : (e - E_);
    for (int hd = 0; hd < heads; ++hd) {
        float am = amax[dst * heads + hd];
        if (am < -3.0e38f) am = 0.f;              // where(isfinite, amax, 0)
        float v = __expf(alpha[(size_t)e * heads + hd] - am);
        exv[(size_t)e * heads + hd] = v;
        atomicAdd(&denom[dst * heads + hd], v);
    }
}

__global__ __launch_bounds__(256)
void k_agg(const int* __restrict__ ei, const float* __restrict__ H,
           const float* __restrict__ exv, const float* __restrict__ denom,
           float* __restrict__ out, int E_, int n, int heads) {
    int e = blockIdx.x;
    int F = heads * 256;
    int src, dst;
    if (e < E_) { src = ei[e]; dst = ei[E_ + e]; }
    else        { src = dst = e - E_; }
    for (int f = threadIdx.x; f < F; f += 256) {
        int hd = f >> 8;
        float w = exv[(size_t)e * heads + hd] / (denom[dst * heads + hd] + 1e-16f);
        atomicAdd(&out[(size_t)dst * F + f], H[(size_t)src * F + f] * w);
    }
}

// ---------------- bias + LayerNorm + ReLU, emits fp32 + bf16 ----------------
__global__ __launch_bounds__(256)
void k_ln(const float* __restrict__ in, const float* bias, const float* g, const float* b,
          float* __restrict__ out_f32, __bf16* __restrict__ out_bf, int n, int F) {
    int gw   = (blockIdx.x * 256 + threadIdx.x) >> 5;
    int lane = threadIdx.x & 31;
    if (gw >= n) return;
    const float* row = in + (size_t)gw * F;
    float s = 0.f;
    for (int f = lane; f < F; f += 32) s += row[f] + bias[f];
    s = waveReduceSum(s);
    s = __shfl(s, 0, 32);
    float mu = s / (float)F;
    float var = 0.f;
    for (int f = lane; f < F; f += 32) { float d = row[f] + bias[f] - mu; var += d * d; }
    var = waveReduceSum(var);
    var = __shfl(var, 0, 32);
    float inv = rsqrtf(var / (float)F + LN_EPS);
    for (int f = lane; f < F; f += 32) {
        float y = (row[f] + bias[f] - mu) * inv * g[f] + b[f];
        y = fmaxf(y, 0.f);
        out_f32[(size_t)gw * F + f] = y;
        out_bf [(size_t)gw * F + f] = (__bf16)y;
    }
}

// ---------------- mean over nodes + classifier ----------------
__global__ void k_colmean(const float* __restrict__ h, float* gvec, int n, int F) {
    int f = blockIdx.x;
    float a = 0.f;
    for (int i = threadIdx.x; i < n; i += 256) a += h[(size_t)i * F + f];
    __shared__ float sh[256];
    sh[threadIdx.x] = a; __syncthreads();
    for (int st = 128; st > 0; st >>= 1) {
        if (threadIdx.x < st) sh[threadIdx.x] += sh[threadIdx.x + st];
        __syncthreads();
    }
    if (threadIdx.x == 0) gvec[f] = sh[0] / (float)n;
}

__global__ void k_cls(const float* gvec, const float* w1, const float* b1,
                      const float* w2, const float* b2, float* out) {
    __shared__ float c1[128];
    int t = threadIdx.x;
    if (t < 128) {
        float a = b1[t];
        for (int i = 0; i < 256; ++i) a += gvec[i] * w1[i * 128 + t];
        c1[t] = fmaxf(a, 0.f);
    }
    __syncthreads();
    if (t < 2) {
        float a = b2[t];
        for (int i = 0; i < 128; ++i) a += c1[i] * w2[i * 2 + t];
        out[t] = a;
    }
}

// ======================================================================
extern "C" void kernel_launch(void* const* d_in, const int* in_sizes, int n_in,
                              void* d_out, int out_size, void* d_ws, size_t ws_size,
                              hipStream_t stream) {
    const float* x   = (const float*)d_in[0];
    const int*   ei  = (const int*)  d_in[1];
    const float* ea  = (const float*)d_in[2];
    const float* qw  = (const float*)d_in[3];
    const float* qb  = (const float*)d_in[4];
    const float* kw  = (const float*)d_in[5];
    const float* kb  = (const float*)d_in[6];
    const float* vw  = (const float*)d_in[7];
    const float* vb  = (const float*)d_in[8];
    const float* ow  = (const float*)d_in[9];
    const float* ob  = (const float*)d_in[10];

    const float* Wl[3]   = {(const float*)d_in[11], (const float*)d_in[19], (const float*)d_in[27]};
    const float* Wel[3]  = {(const float*)d_in[12], (const float*)d_in[20], (const float*)d_in[28]};
    const float* asl[3]  = {(const float*)d_in[13], (const float*)d_in[21], (const float*)d_in[29]};
    const float* adl[3]  = {(const float*)d_in[14], (const float*)d_in[22], (const float*)d_in[30]};
    const float* ael[3]  = {(const float*)d_in[15], (const float*)d_in[23], (const float*)d_in[31]};
    const float* bl[3]   = {(const float*)d_in[16], (const float*)d_in[24], (const float*)d_in[32]};
    const float* lngl[3] = {(const float*)d_in[17], (const float*)d_in[25], (const float*)d_in[33]};
    const float* lnbl[3] = {(const float*)d_in[18], (const float*)d_in[26], (const float*)d_in[34]};
    const float* cw1 = (const float*)d_in[35];
    const float* cb1 = (const float*)d_in[36];
    const float* cw2 = (const float*)d_in[37];
    const float* cb2 = (const float*)d_in[38];

    const int Dl[3] = {1280, 1024, 1024};
    const int Fl[3] = {1024, 1024, 256};
    const int Hd[3] = {4, 4, 1};

    const int nN = in_sizes[0] / (EMB + MDIM);   // 20000
    const int E_ = in_sizes[2];                  // 200000
    const int TOT = E_ + nN;

    // ---- workspace carve ----
    char* wp = (char*)d_ws;
    auto carve = [&](size_t bytes) -> void* {
        void* p = (void*)wp;
        wp += (bytes + 255) & ~(size_t)255;
        return p;
    };
    float*  h0    = (float*) carve((size_t)nN * EMB  * 4); // layer input fp32
    float*  Hbuf  = (float*) carve((size_t)nN * 1024 * 4); // projected features
    float*  h2    = (float*) carve((size_t)nN * 1024 * 4); // aggregated output
    __bf16* hb    = (__bf16*)carve((size_t)nN * EMB  * 2); // layer input bf16
    __bf16* wt    = (__bf16*)carve((size_t)1280 * 1024 * 2);
    float*  Kvec  = (float*) carve(256 * 4);
    float*  Vvec  = (float*) carve(256 * 4);
    float*  qbK   = (float*) carve(4);
    float*  wvec  = (float*) carve(EMB * 4);
    float*  uvec  = (float*) carve(EMB * 4);
    float*  sco   = (float*) carve((size_t)nN * 4);
    float*  exn   = (float*) carve((size_t)nN * 4);
    float*  part  = (float*) carve(256 * 4);
    float*  gmaxv = (float*) carve(4);
    float*  gsumv = (float*) carve(4);
    float*  cnt   = (float*) carve((size_t)nN * 4);
    float*  sattr = (float*) carve((size_t)nN * 4);
    float*  loopA = (float*) carve((size_t)nN * 4);
    float*  al_s  = (float*) carve((size_t)nN * 4 * 4);
    float*  al_d  = (float*) carve((size_t)nN * 4 * 4);
    float*  wedot = (float*) carve(4 * 4);
    float*  amax  = (float*) carve((size_t)nN * 4 * 4);
    float*  den   = (float*) carve((size_t)nN * 4 * 4);
    float*  alph  = (float*) carve((size_t)TOT * 4 * 4);
    float*  exv   = (float*) carve((size_t)TOT * 4 * 4);
    float*  gvec  = (float*) carve(256 * 4);

    auto cdiv = [](long long a, long long b) { return (int)((a + b - 1) / b); };

    // ---- stage 0: degrees + self-loop attr (constant across layers) ----
    k_fill<<<cdiv(nN,256),256,0,stream>>>(cnt,   (size_t)nN, 0.f);
    k_fill<<<cdiv(nN,256),256,0,stream>>>(sattr, (size_t)nN, 0.f);
    k_deg <<<cdiv(E_,256),256,0,stream>>>(ei, ea, E_, cnt, sattr);
    k_loopattr<<<cdiv(nN,256),256,0,stream>>>(cnt, sattr, loopA, nN);

    // ---- stage 1: cross-attention (rank-1 collapsed) ----
    k_kv<<<1,256,0,stream>>>(x, kw, kb, vw, vb, qb, Kvec, Vvec, qbK);
    k_wu<<<cdiv(EMB,256),256,0,stream>>>(qw, ow, Kvec, Vvec, wvec, uvec);
    k_scores<<<cdiv((long long)nN*32,256),256,0,stream>>>(x, wvec, qbK, sco, nN);
    k_pmax<<<256,256,0,stream>>>(sco, nN, part);
    k_gmax<<<1,256,0,stream>>>(part, gmaxv);
    k_expsum<<<256,256,0,stream>>>(sco, gmaxv, nN, exn, part);
    k_gsum<<<1,256,0,stream>>>(part, gsumv);
    k_h0<<<cdiv((long long)nN*EMB,256),256,0,stream>>>(x, ob, uvec, exn, gsumv, h0, hb, nN);

    // ---- stage 2: 3 GAT layers ----
    for (int l = 0; l < 3; ++l) {
        const int D = Dl[l], F = Fl[l], heads = Hd[l];

        // weights -> bf16 transposed
        k_wtrans<<<cdiv((long long)D*F,256),256,0,stream>>>(Wl[l], wt, D, F);

        // H = hb @ W   (WMMA bf16, fp32 accumulate)
        dim3 ggrid(F / 128, cdiv(nN, 128));
        k_gemm_bf16<<<ggrid,256,0,stream>>>(hb, wt, Hbuf, nN, F, D);

        // attention logits
        k_aldot<<<cdiv((long long)nN*heads*32,256),256,0,stream>>>(Hbuf, asl[l], adl[l], al_s, al_d, nN, heads);
        k_wedot<<<1,256,0,stream>>>(Wel[l], ael[l], wedot, heads);

        k_fill<<<cdiv((long long)nN*heads,256),256,0,stream>>>(amax, (size_t)nN*heads, -INFINITY);
        k_fill<<<cdiv((long long)nN*heads,256),256,0,stream>>>(den,  (size_t)nN*heads, 0.f);
        k_alpha<<<cdiv(TOT,256),256,0,stream>>>(ei, ea, loopA, al_s, al_d, wedot, alph, amax, E_, nN, heads);
        k_exden<<<cdiv(TOT,256),256,0,stream>>>(ei, alph, amax, exv, den, E_, nN, heads);

        // aggregate
        k_fill<<<cdiv((long long)nN*F,256),256,0,stream>>>(h2, (size_t)nN*F, 0.f);
        k_agg<<<TOT,256,0,stream>>>(ei, Hbuf, exv, den, h2, E_, nN, heads);

        // bias + LN + ReLU -> next layer input (fp32 + bf16)
        k_ln<<<cdiv((long long)nN*32,256),256,0,stream>>>(h2, bl[l], lngl[l], lnbl[l], h0, hb, nN, F);
    }

    // ---- stage 3: mean pool + classifier ----
    k_colmean<<<256,256,0,stream>>>(h0, gvec, nN, 256);
    k_cls<<<1,256,0,stream>>>(gvec, cw1, cb1, cw2, cb2, (float*)d_out);
}
